// LightGCN_27058293964948
// MI455X (gfx1250) — compile-verified, matrix operand order checked by reading, not run on hardware
//
#include <hip/hip_runtime.h>
#include <hip/hip_bf16.h>
#include <math.h>

#define NUM_USER  100000
#define NUM_ITEM  50000
#define HID       64
#define NNZ2      3200000        // 2 * 1,600,000 symmetrized edges
#define N_NODES   (NUM_USER + NUM_ITEM)
#define N_LAYERS  3
#define REG_C     0.0001f
#define BATCH     16384

typedef __attribute__((ext_vector_type(2)))  float    v2f;
typedef __attribute__((ext_vector_type(8)))  float    v8f;
typedef __attribute__((ext_vector_type(16))) _Float16 v16h;

// ---------------------------------------------------------------------------
// x = concat(user_emb, item_emb); acc = x; d_out = 0
// ---------------------------------------------------------------------------
__global__ void k_init(const float* __restrict__ ue, const float* __restrict__ ie,
                       float* __restrict__ x, float* __restrict__ acc,
                       float* __restrict__ out) {
    size_t idx = (size_t)blockIdx.x * blockDim.x + threadIdx.x;   // float4 units
    if (idx == 0) out[0] = 0.0f;
    const size_t NH4 = (size_t)N_NODES * HID / 4;
    const size_t UH4 = (size_t)NUM_USER * HID / 4;
    if (idx < NH4) {
        float4 v = (idx < UH4) ? ((const float4*)ue)[idx]
                               : ((const float4*)ie)[idx - UH4];
        ((float4*)x)[idx]   = v;
        ((float4*)acc)[idx] = v;
    }
}

__global__ void k_zero(float* __restrict__ y) {
    size_t idx = (size_t)blockIdx.x * blockDim.x + threadIdx.x;
    const size_t NH4 = (size_t)N_NODES * HID / 4;
    if (idx < NH4) ((float4*)y)[idx] = make_float4(0.f, 0.f, 0.f, 0.f);
}

// ---------------------------------------------------------------------------
// One wave per edge (32 lanes x float2 = 64 dims).  x table (38.4 MB) is
// L2-resident on MI455X (192 MB L2), so gathers + scatter atomics stay on-chip.
// ---------------------------------------------------------------------------
__global__ void k_scatter(const int* __restrict__ er, const int* __restrict__ ec,
                          const float* __restrict__ ev,
                          const float* __restrict__ x, float* __restrict__ y) {
    size_t tid = (size_t)blockIdx.x * blockDim.x + threadIdx.x;
    size_t e   = tid >> 5;
    if (e >= (size_t)NNZ2) return;
    const int d2 = (int)(tid & 31) * 2;
    const int r  = er[e];
    const int c  = ec[e];
    const float v = ev[e];
    const float2 xv = *(const float2*)(x + (size_t)c * HID + d2);
    float* dst = y + (size_t)r * HID + d2;
    atomicAdd(dst,     v * xv.x);
    atomicAdd(dst + 1, v * xv.y);
}

__global__ void k_accum(const float* __restrict__ y, float* __restrict__ acc) {
    size_t idx = (size_t)blockIdx.x * blockDim.x + threadIdx.x;
    const size_t NH4 = (size_t)N_NODES * HID / 4;
    if (idx < NH4) {
        float4 a = ((const float4*)acc)[idx];
        float4 b = ((const float4*)y)[idx];
        a.x += b.x; a.y += b.y; a.z += b.z; a.w += b.w;
        ((float4*)acc)[idx] = a;
    }
}

// ---------------------------------------------------------------------------
// BPR scoring: one wave per 16 batch rows.
//   S_p = U(16x64) x P^T,  S_n = U x N^T   via V_WMMA_F32_16X16X4_F32
// Diagonals extracted through LDS; L2 term accumulated from the fragments
// (each matrix element is loaded exactly once across the 16 K-chunks).
// ---------------------------------------------------------------------------
__global__ __launch_bounds__(32)
void k_score(const int* __restrict__ u, const int* __restrict__ i,
             const int* __restrict__ j, const float* __restrict__ acc,
             float* __restrict__ out) {
    __shared__ float lds[2][8][32];
    const int lane = threadIdx.x;
    const int row  = lane & 15;
    const int koff = (lane >> 4) * 2;          // lanes 0-15 -> K+{0,1}, 16-31 -> K+{2,3}
    const int b    = blockIdx.x * 16 + row;

    const float* ub = acc + (size_t)u[b] * HID;
    const float* pb = acc + ((size_t)NUM_USER + i[b]) * HID;
    const float* nb = acc + ((size_t)NUM_USER + j[b]) * HID;

    v8f cp = {};
    v8f cn = {};
    float ssq = 0.0f;

#if __has_builtin(__builtin_amdgcn_wmma_f32_16x16x4_f32)
    #pragma unroll
    for (int t = 0; t < 16; ++t) {
        const int kb = t * 4 + koff;
        v2f a  = *(const v2f*)(ub + kb);
        v2f bp = *(const v2f*)(pb + kb);
        v2f bn = *(const v2f*)(nb + kb);
        ssq += a.x * a.x + a.y * a.y + bp.x * bp.x + bp.y * bp.y
             + bn.x * bn.x + bn.y * bn.y;
        cp = __builtin_amdgcn_wmma_f32_16x16x4_f32(false, a, false, bp,
                                                   (short)0, cp, false, false);
        cn = __builtin_amdgcn_wmma_f32_16x16x4_f32(false, a, false, bn,
                                                   (short)0, cn, false, false);
    }
#else
    // Fallback: f16 inputs / f32 accumulate (codegen-confirmed builtin).
    // 16-bit A 16x32, contiguous K chosen here (A and B alike): lane holds row (lane&15);
    // lanes 0-15 take K {0..15}, lanes 16-31 take {16..31}; two 32-K chunks cover HID=64.
    #pragma unroll
    for (int half = 0; half < 2; ++half) {
        v16h a = {}, bp16 = {}, bn16 = {};
        const int kbase = half * 32 + (lane >> 4) * 16;  // first 16 K of this lane half
        #pragma unroll
        for (int q = 0; q < 16; ++q) {
            float av = ub[kbase + q], pv = pb[kbase + q], nv = nb[kbase + q];
            a[q]    = (_Float16)av;
            bp16[q] = (_Float16)pv;
            bn16[q] = (_Float16)nv;
            ssq += av * av + pv * pv + nv * nv;
        }
        cp = __builtin_amdgcn_wmma_f32_16x16x32_f16(false, a, false, bp16,
                                                    (short)0, cp, false, false);
        cn = __builtin_amdgcn_wmma_f32_16x16x32_f16(false, a, false, bn16,
                                                    (short)0, cn, false, false);
    }
#endif

    #pragma unroll
    for (int r = 0; r < 8; ++r) {
        lds[0][r][lane] = cp[r];
        lds[1][r][lane] = cn[r];
    }
    __syncthreads();

    // final = acc / (N_LAYERS+1): both operands carried 1/4 implicitly -> 1/16.
    float lloss = 0.0f;
    if (lane < 16) {
        const float s  = 0.0625f;
        const float dp = (lane < 8) ? lds[0][lane][lane] : lds[0][lane - 8][lane + 16];
        const float dn = (lane < 8) ? lds[1][lane][lane] : lds[1][lane - 8][lane + 16];
        const float d  = s * (dp - dn);
        // softplus(-d) = -log_sigmoid(d), numerically stable
        lloss = (d > 0.0f) ? log1pf(expf(-d)) : (-d + log1pf(expf(d)));
    }

    // wave32 reductions
    #pragma unroll
    for (int m = 16; m >= 1; m >>= 1) {
        lloss += __shfl_xor(lloss, m, 32);
        ssq   += __shfl_xor(ssq,   m, 32);
    }
    if (lane == 0) {
        // ssq was on un-scaled acc rows: scale by (1/4)^2; L2 = ssq/2/BATCH
        const float contrib = lloss / (float)BATCH
                            + REG_C * (ssq * 0.0625f) * 0.5f / (float)BATCH;
        atomicAdd(out, contrib);
    }
}

// ---------------------------------------------------------------------------
extern "C" void kernel_launch(void* const* d_in, const int* in_sizes, int n_in,
                              void* d_out, int out_size, void* d_ws, size_t ws_size,
                              hipStream_t stream) {
    const int*   u  = (const int*)d_in[0];
    const int*   i  = (const int*)d_in[1];
    const int*   j  = (const int*)d_in[2];
    const int*   er = (const int*)d_in[3];
    const int*   ec = (const int*)d_in[4];
    const float* ev = (const float*)d_in[5];
    const float* ue = (const float*)d_in[6];
    const float* ie = (const float*)d_in[7];
    float* out = (float*)d_out;

    const size_t NH = (size_t)N_NODES * HID;      // 9.6M floats = 38.4 MB
    float* x   = (float*)d_ws;
    float* y   = x + NH;
    float* acc = y + NH;

    const int T = 256;
    const unsigned gElem = (unsigned)((NH / 4 + T - 1) / T);
    k_init<<<gElem, T, 0, stream>>>(ue, ie, x, acc, out);

    const unsigned gScat = (unsigned)(((size_t)NNZ2 * 32 + T - 1) / T);
    for (int l = 0; l < N_LAYERS; ++l) {
        k_zero<<<gElem, T, 0, stream>>>(y);
        k_scatter<<<gScat, T, 0, stream>>>(er, ec, ev, x, y);
        k_accum<<<gElem, T, 0, stream>>>(y, acc);
        float* tmp = x; x = y; y = tmp;
    }

    k_score<<<BATCH / 16, 32, 0, stream>>>(u, i, j, acc, out);
}
